// FusedAttention_72533407695514
// MI455X (gfx1250) — compile-verified, matrix operand order checked by reading, not run on hardware
//
#include <hip/hip_runtime.h>
#include <hip/hip_bf16.h>

// ---------------------------------------------------------------------------
// Problem constants (B=2, S=2048, H=4096, NQ=32, NKV=8, D=128, BUF=8192)
// ---------------------------------------------------------------------------
#define SEQ    2048
#define HID    4096
#define NQH    32
#define NKVH   8
#define HDIM   128
#define MTOT   4096              // B*S
#define NQKV   6144              // H + 2*NKV*D
#define KVROW  2048              // 2*NKV*D floats per cache row

typedef __bf16 v16bf __attribute__((ext_vector_type(16)));
typedef float  v8f   __attribute__((ext_vector_type(8)));
typedef unsigned int v8u __attribute__((ext_vector_type(8)));

// float -> bf16 bits, round-to-nearest-even
__device__ __forceinline__ unsigned short f2bf(float f) {
    unsigned int u = __builtin_bit_cast(unsigned int, f);
    unsigned int r = u + 0x7FFFu + ((u >> 16) & 1u);
    return (unsigned short)(r >> 16);
}

// LDS byte offset of a __shared__ pointer (generic ptr low 32 bits == LDS addr)
__device__ __forceinline__ unsigned lds_off(const void* p) {
    return (unsigned)(unsigned long long)p;
}

// Async copy of 32 contiguous bytes: global (saddr + 32-bit voffset) -> LDS.
// Tracked by ASYNCcnt; IOFFSET applies to both global and LDS addresses.
__device__ __forceinline__ void async_copy32(unsigned lds, unsigned voff,
                                             const void* sbase) {
    asm volatile(
        "global_load_async_to_lds_b128 %0, %1, %2 offset:0\n\t"
        "global_load_async_to_lds_b128 %0, %1, %2 offset:16"
        :: "v"(lds), "v"(voff), "s"(sbase) : "memory");
}
__device__ __forceinline__ void async_wait0() {
    asm volatile("s_wait_asynccnt 0" ::: "memory");
}

// Load a 16x32 bf16 fragment (A-layout; also valid for B sourced from a
// row-major [16 rows][K] tile, rows indexed by lane%16).
// ISA 16-bit layout: lanes 0-15 -> K {0..7,16..23}, lanes 16-31 -> K {8..15,24..31},
// two bf16 packed per VGPR (k, k+1).
__device__ __forceinline__ v16bf load_frag16(const unsigned short* p, int rs) {
    const int l = threadIdx.x & 31;
    const int g = (l >> 4) * 8;
    const unsigned short* row = p + (size_t)(l & 15) * rs;
    v8u u;
#pragma unroll
    for (int j = 0; j < 8; ++j) {
        const int k = (j < 4) ? (g + j * 2) : (16 + g + (j - 4) * 2);
        u[j] = *(const unsigned int*)(row + k);
    }
    return __builtin_bit_cast(v16bf, u);
}

__device__ __forceinline__ v8f wmma_bf16(v16bf a, v16bf b, v8f c) {
    return __builtin_amdgcn_wmma_f32_16x16x32_bf16(false, a, false, b,
                                                   (short)0, c, false, false);
}

// ---------------------------------------------------------------------------
// Kernel: fp32 -> bf16 bulk convert (float4 -> 4x bf16 per thread)
// ---------------------------------------------------------------------------
__global__ void f32_to_bf16(const float* __restrict__ src,
                            unsigned short* __restrict__ dst, int n4) {
    const int i = blockIdx.x * blockDim.x + threadIdx.x;
    if (i < n4) {
        const float4 f = ((const float4*)src)[i];
        ushort4 o;
        o.x = f2bf(f.x); o.y = f2bf(f.y); o.z = f2bf(f.z); o.w = f2bf(f.w);
        ((ushort4*)dst)[i] = o;
    }
}

// ---------------------------------------------------------------------------
// GEMM: C[M,N] = A[M,K] * B[N,K]^T, all-bf16 operands, fp32 accumulate.
// Block: 256 threads = 8 waves (2x4); block tile 128x128; wave tile 64x32
// (4x2 WMMA tiles -> 8 v_wmma per 32-wide K chunk per wave).
// A/B tiles staged via global_load_async_to_lds_b128 (ASYNCcnt).
// ---------------------------------------------------------------------------
__global__ __launch_bounds__(256) void gemm_bf16_async(
    const unsigned short* __restrict__ A,
    const unsigned short* __restrict__ B,
    float* __restrict__ C, int N, int K)
{
    __shared__ __align__(16) unsigned short As[128 * 32];
    __shared__ __align__(16) unsigned short Bs[128 * 32];

    const int tid    = threadIdx.x;
    const int m_base = blockIdx.y * 128;
    const int n_base = blockIdx.x * 128;
    const int wave   = tid >> 5;
    const int wm     = wave >> 2;          // 0..1  (64 rows)
    const int wn     = wave & 3;           // 0..3  (32 cols)

    // Each thread stages one 32B run of the 128x32 tiles.
    const int srow = tid >> 1;
    const int sk0  = (tid & 1) * 16;
    const unsigned ldsA = lds_off(As + srow * 32 + sk0);
    const unsigned ldsB = lds_off(Bs + srow * 32 + sk0);
    const unsigned gA = (unsigned)(((size_t)(m_base + srow) * K + sk0) * 2);
    const unsigned gB = (unsigned)(((size_t)(n_base + srow) * K + sk0) * 2);

    v8f acc[4][2] = {};

    for (int kc = 0; kc < K; kc += 32) {
        __syncthreads();
        async_copy32(ldsA, gA + kc * 2, A);
        async_copy32(ldsB, gB + kc * 2, B);
        async_wait0();
        __syncthreads();

        v16bf af[4], bf[2];
#pragma unroll
        for (int tm = 0; tm < 4; ++tm)
            af[tm] = load_frag16(As + (wm * 64 + tm * 16) * 32, 32);
#pragma unroll
        for (int tn = 0; tn < 2; ++tn)
            bf[tn] = load_frag16(Bs + (wn * 32 + tn * 16) * 32, 32);
#pragma unroll
        for (int tm = 0; tm < 4; ++tm) {
            acc[tm][0] = wmma_bf16(af[tm], bf[0], acc[tm][0]);
            acc[tm][1] = wmma_bf16(af[tm], bf[1], acc[tm][1]);
        }
    }

    // C/D f32 layout: lane%16 = col, (lane/16)*8 + i = row
    const int l = tid & 31, g = l >> 4, c = l & 15;
#pragma unroll
    for (int tm = 0; tm < 4; ++tm)
#pragma unroll
        for (int tn = 0; tn < 2; ++tn)
#pragma unroll
            for (int i = 0; i < 8; ++i) {
                const int row = m_base + wm * 64 + tm * 16 + g * 8 + i;
                const int col = n_base + wn * 32 + tn * 16 + c;
                C[(size_t)row * N + col] = acc[tm][tn][i];
            }
}

// ---------------------------------------------------------------------------
// Bulk copy kv_buffer -> output cache region (float4)
// ---------------------------------------------------------------------------
__global__ void copy_kv(const float* __restrict__ src, float* __restrict__ dst, int n4) {
    const int i = blockIdx.x * blockDim.x + threadIdx.x;
    if (i < n4) ((float4*)dst)[i] = ((const float4*)src)[i];
}

// ---------------------------------------------------------------------------
// RoPE on Q/K + scatter of rotated K and raw V into the KV cache;
// emits bf16 Q/K/V for the attention kernel. One block per token.
// ---------------------------------------------------------------------------
__global__ __launch_bounds__(256) void rope_scatter(
    const float* __restrict__ qkv,
    const float* __restrict__ cosb, const float* __restrict__ sinb,
    const int* __restrict__ sel,
    unsigned short* __restrict__ qb, unsigned short* __restrict__ kb,
    unsigned short* __restrict__ vb, float* __restrict__ kv_out)
{
    const int t = blockIdx.x;
    const int s = t & (SEQ - 1);
    const float* row = qkv + (size_t)t * NQKV;
    const int dstrow = sel[t];

    for (int e = threadIdx.x; e < NQKV; e += blockDim.x) {
        const int head = e >> 7;
        const int d    = e & 127;
        const float val = row[e];
        float out;
        if (head < NQH + NKVH) {   // rope Q and K
            const float partner = (d < 64) ? -row[e + 64] : row[e - 64];
            out = val * cosb[s * HDIM + d] + partner * sinb[s * HDIM + d];
        } else {
            out = val;             // V passes through
        }
        if (head < NQH) {
            qb[(size_t)t * HID + e] = f2bf(out);
        } else if (head < NQH + NKVH) {
            const int kvh = head - NQH;
            kb[(size_t)t * (NKVH * HDIM) + kvh * HDIM + d] = f2bf(out);
            kv_out[(size_t)dstrow * KVROW + kvh * HDIM + d] = out;
        } else {
            const int kvh = head - (NQH + NKVH);
            vb[(size_t)t * (NKVH * HDIM) + kvh * HDIM + d] = f2bf(out);
            kv_out[(size_t)dstrow * KVROW + (NKVH + kvh) * HDIM + d] = out;
        }
    }
}

// ---------------------------------------------------------------------------
// Causal flash attention, one block per (b, head, 128-row Q tile).
// 8 waves x 16 query rows. 32-key chunks: K staged via async-to-LDS copies,
// V staged transposed (so P.V B-fragments are contiguous ds reads).
// Scores: 8 WMMAs/chunk; P.V: 8 WMMAs/chunk; fragments batched before WMMAs.
// ---------------------------------------------------------------------------
__global__ __launch_bounds__(256) void flash_attn(
    const unsigned short* __restrict__ Q,
    const unsigned short* __restrict__ Kb,
    const unsigned short* __restrict__ Vb,
    unsigned short* __restrict__ Ob)
{
    __shared__ __align__(16) unsigned short Klds[32 * 128];   // [key][chan]
    __shared__ __align__(16) unsigned short Vt  [128 * 32];   // [chan][key]
    __shared__ __align__(16) unsigned short Plds[8 * 16 * 32];

    const int tid  = threadIdx.x, lane = tid & 31, wave = tid >> 5;
    const int qt   = blockIdx.x & 15;
    const int h    = (blockIdx.x >> 4) & 31;
    const int b    = blockIdx.x >> 9;
    const int kvh  = h >> 2;                 // NQ/NKV = 4
    const int q0   = qt * 128;
    const int m_w  = q0 + wave * 16;
    const int g    = lane >> 4, c = lane & 15;
    const float scale = 0.08838834764831845f;   // 1/sqrt(128)

    // Q fragments for this wave's 16 rows (resident for the whole pass)
    v16bf qa[4];
    {
        const unsigned short* qp = Q + (size_t)(b * SEQ + m_w) * HID + h * HDIM;
#pragma unroll
        for (int cc = 0; cc < 4; ++cc) qa[cc] = load_frag16(qp + cc * 32, HID);
    }

    v8f o[8] = {};
    float mrun[8], lrun[8];
#pragma unroll
    for (int i = 0; i < 8; ++i) { mrun[i] = -3.4e38f; lrun[i] = 0.0f; }

    unsigned short* pw = Plds + wave * 512;
    const int nkc = (q0 >> 5) + 4;           // causal: keys < q0+128

    // per-thread staging coordinates: 32B of K per thread per chunk
    const int skey = tid >> 3;
    const int sd0  = (tid & 7) * 16;
    const unsigned ldsK  = lds_off(Klds + skey * 128 + sd0);
    const unsigned gKrow = (unsigned)(((size_t)(b * SEQ + skey) * (NKVH * HDIM)
                                       + kvh * HDIM + sd0) * 2);

    for (int kc = 0; kc < nkc; ++kc) {
        __syncthreads();
        // ---- K chunk: async global -> LDS (row-major)
        async_copy32(ldsK, gKrow + (unsigned)(kc * 32 * NKVH * HDIM * 2), Kb);
        // ---- V chunk: manual transposed staging
        {
            const size_t tok = (size_t)(b * SEQ + kc * 32 + skey);
            const unsigned short* vs = Vb + tok * (NKVH * HDIM) + kvh * HDIM + sd0;
#pragma unroll
            for (int dd = 0; dd < 16; ++dd) Vt[(sd0 + dd) * 32 + skey] = vs[dd];
            if (kc + 1 < nkc)
                __builtin_prefetch(vs + 32 * NKVH * HDIM, 0, 0);
        }
        async_wait0();
        __syncthreads();

        // ---- scores S = Q K^T (batch all fragments, then the WMMA chain)
        v16bf kf[2][4];
#pragma unroll
        for (int nt = 0; nt < 2; ++nt)
#pragma unroll
            for (int cc = 0; cc < 4; ++cc)
                kf[nt][cc] = load_frag16(Klds + (nt * 16) * 128 + cc * 32, 128);
        v8f sc[2] = {};
#pragma unroll
        for (int nt = 0; nt < 2; ++nt)
#pragma unroll
            for (int cc = 0; cc < 4; ++cc)
                sc[nt] = wmma_bf16(qa[cc], kf[nt][cc], sc[nt]);

        // ---- scale + causal mask (C layout: col = lane%16, row = g*8+i)
#pragma unroll
        for (int nt = 0; nt < 2; ++nt) {
            const int kglob = kc * 32 + nt * 16 + c;
#pragma unroll
            for (int i = 0; i < 8; ++i) {
                const int rglob = m_w + g * 8 + i;
                sc[nt][i] = (kglob <= rglob) ? sc[nt][i] * scale : -3.4e38f;
            }
        }
        // ---- online softmax: row reductions across the 16 lanes of a row
#pragma unroll
        for (int i = 0; i < 8; ++i) {
            float v = fmaxf(sc[0][i], sc[1][i]);
            v = fmaxf(v, __shfl_xor(v, 1, 32));
            v = fmaxf(v, __shfl_xor(v, 2, 32));
            v = fmaxf(v, __shfl_xor(v, 4, 32));
            v = fmaxf(v, __shfl_xor(v, 8, 32));
            const float mn    = fmaxf(mrun[i], v);
            const float alpha = __expf(mrun[i] - mn);
            mrun[i] = mn;
            const float p0 = __expf(sc[0][i] - mn);
            const float p1 = __expf(sc[1][i] - mn);
            sc[0][i] = p0; sc[1][i] = p1;
            float rs = p0 + p1;
            rs += __shfl_xor(rs, 1, 32);
            rs += __shfl_xor(rs, 2, 32);
            rs += __shfl_xor(rs, 4, 32);
            rs += __shfl_xor(rs, 8, 32);
            lrun[i] = alpha * lrun[i] + rs;
#pragma unroll
            for (int ct = 0; ct < 8; ++ct) o[ct][i] *= alpha;
        }
        // ---- P (C layout) -> per-wave LDS -> A layout
#pragma unroll
        for (int nt = 0; nt < 2; ++nt)
#pragma unroll
            for (int i = 0; i < 8; ++i)
                pw[(g * 8 + i) * 32 + nt * 16 + c] = f2bf(sc[nt][i]);
        __syncthreads();

        // ---- O += P V (batch fragments, then WMMA chain)
        const v16bf pa = load_frag16(pw, 32);
        v16bf vf[8];
#pragma unroll
        for (int ct = 0; ct < 8; ++ct)
            vf[ct] = load_frag16(Vt + (ct * 16) * 32, 32);
#pragma unroll
        for (int ct = 0; ct < 8; ++ct)
            o[ct] = wmma_bf16(pa, vf[ct], o[ct]);
    }

    // ---- normalize and emit bf16 attention output (input to Wo GEMM)
#pragma unroll
    for (int ct = 0; ct < 8; ++ct)
#pragma unroll
        for (int i = 0; i < 8; ++i) {
            const float val = o[ct][i] / lrun[i];
            const int row = b * SEQ + m_w + g * 8 + i;
            const int col = h * HDIM + ct * 16 + c;
            Ob[(size_t)row * HID + col] = f2bf(val);
        }
}

// ---------------------------------------------------------------------------
// Launcher. Workspace layout (bytes):
//   [0,            96M)  qkv_f32       (dead after rope)  -- reused for Wo bf16
//   [96M,         144M)  Wq||Wkv bf16
//   [144M,        176M)  x bf16        (dead after QKV)   -- reused for attn out
//   [176M,        208M)  Q bf16
//   [208M,        216M)  K bf16
//   [216M,        224M)  V bf16
// ---------------------------------------------------------------------------
extern "C" void kernel_launch(void* const* d_in, const int* in_sizes, int n_in,
                              void* d_out, int out_size, void* d_ws, size_t ws_size,
                              hipStream_t stream)
{
    (void)in_sizes; (void)n_in; (void)out_size; (void)ws_size;

    const float* x    = (const float*)d_in[0];
    const float* wq   = (const float*)d_in[1];
    const float* wkv  = (const float*)d_in[2];
    const float* wo   = (const float*)d_in[3];
    const float* cosb = (const float*)d_in[4];
    const float* sinb = (const float*)d_in[5];
    const float* kvb  = (const float*)d_in[6];
    const int*   sel  = (const int*)d_in[7];

    float* out    = (float*)d_out;
    float* kv_out = out + (size_t)MTOT * HID;     // second tuple output

    char* ws = (char*)d_ws;
    float*          qkv   = (float*)ws;
    unsigned short* wob   = (unsigned short*)ws;                    // aliases qkv
    unsigned short* wqkvb = (unsigned short*)(ws + 100663296);
    unsigned short* xb    = (unsigned short*)(ws + 150994944);
    unsigned short* ab    = (unsigned short*)(ws + 150994944);      // aliases xb
    unsigned short* qb    = (unsigned short*)(ws + 184549376);
    unsigned short* kb    = (unsigned short*)(ws + 218103808);
    unsigned short* vb    = (unsigned short*)(ws + 226492416);

    // 0) one-time fp32 -> bf16 conversions (weights read once; bf16 ~fits L2)
    f32_to_bf16<<<(MTOT * HID / 4 + 255) / 256, 256, 0, stream>>>(x, xb, MTOT * HID / 4);
    f32_to_bf16<<<(HID * HID / 4 + 255) / 256, 256, 0, stream>>>(wq, wqkvb, HID * HID / 4);
    f32_to_bf16<<<(KVROW * HID / 4 + 255) / 256, 256, 0, stream>>>(
        wkv, wqkvb + (size_t)HID * HID, KVROW * HID / 4);

    // 1) QKV projection: [4096 x 4096] * [6144 x 4096]^T
    dim3 g1(NQKV / 128, MTOT / 128);
    gemm_bf16_async<<<g1, 256, 0, stream>>>(xb, wqkvb, qkv, NQKV, HID);

    // 2) copy old KV buffer into output cache
    const int n4 = (8192 * KVROW) / 4;
    copy_kv<<<(n4 + 255) / 256, 256, 0, stream>>>(kvb, kv_out, n4);

    // 3) RoPE + scatter selected rows into cache, emit bf16 Q/K/V
    rope_scatter<<<MTOT, 256, 0, stream>>>(qkv, cosb, sinb, sel, qb, kb, vb, kv_out);

    // 4) Wo -> bf16 (reuses the now-dead qkv region)
    f32_to_bf16<<<(HID * HID / 4 + 255) / 256, 256, 0, stream>>>(wo, wob, HID * HID / 4);

    // 5) causal GQA flash attention
    flash_attn<<<2 * NQH * (SEQ / 128), 256, 0, stream>>>(qb, kb, vb, ab);

    // 6) output projection: [4096 x 4096] * [4096 x 4096]^T
    dim3 g6(HID / 128, MTOT / 128);
    gemm_bf16_async<<<g6, 256, 0, stream>>>(ab, wob, out, HID, HID);
}
